// Self_nysAttn_35313221107721
// MI455X (gfx1250) — matrix operand order repaired and checked by reading, not compiled, not run on hardware
//
#include <hip/hip_runtime.h>
#include <hip/hip_bf16.h>
#include <math.h>

typedef _Float16 half_t;
typedef __attribute__((ext_vector_type(16))) _Float16 v16h;
typedef __attribute__((ext_vector_type(8)))  _Float16 v8h;
typedef __attribute__((ext_vector_type(8)))  float    v8f;

#define B_   2
#define C_   64
#define N_   4096
#define D_   8
#define M_   256
#define L_   16
#define NS_ITERS 6
#define SCALE_ 0.35355339059327373f  /* 1/sqrt(8) */

// ---------------------------------------------------------------------------
// WMMA fragment loaders (layouts per CDNA5 ISA §7.12.2, wave32)
// ---------------------------------------------------------------------------
// A: 16x32 f16, row-major source. lanes 0-15: row=lane, K=0..7 & 16..23;
//    lanes 16-31: row=lane-16, K=8..15 & 24..31.
__device__ __forceinline__ v16h load_a_frag(const half_t* __restrict__ A, int lda,
                                            int row0, int k0, int lane) {
  int r  = lane & 15;
  int kb = (lane & 16) ? 8 : 0;
  const half_t* p = A + (size_t)(row0 + r) * lda + k0 + kb;
  v8h lo = *(const v8h*)(p);
  v8h hi = *(const v8h*)(p + 16);
  v16h out;
#pragma unroll
  for (int i = 0; i < 8; i++) { out[i] = lo[i]; out[i + 8] = hi[i]; }
  return out;
}

// Same A fragment but converting from an f32 row-major source (attn matrix).
__device__ __forceinline__ v16h load_a_frag_f32(const float* __restrict__ A, int lda,
                                                int row0, int k0, int lane) {
  int r  = lane & 15;
  int kb = (lane & 16) ? 8 : 0;
  const float* p = A + (size_t)(row0 + r) * lda + k0 + kb;
  v16h out;
#pragma unroll
  for (int i = 0; i < 8; i++) { out[i] = (half_t)p[i]; out[i + 8] = (half_t)p[i + 16]; }
  return out;
}

// B: 32x16 f16 from a TRANSPOSED [cols][K] row-major array.
// lanes 0-15: col=lane, K=0..7 & 16..23; lanes 16-31: col=lane-16, K=8..15 & 24..31.
__device__ __forceinline__ v16h load_b_frag(const half_t* __restrict__ Bt, int ldb,
                                            int col0, int k0, int lane) {
  int n  = lane & 15;
  int ks = (lane & 16) ? 8 : 0;
  const half_t* p = Bt + (size_t)(col0 + n) * ldb + k0 + ks;
  v8h lo = *(const v8h*)(p);
  v8h hi = *(const v8h*)(p + 16);
  v16h out;
#pragma unroll
  for (int i = 0; i < 8; i++) { out[i] = lo[i]; out[i + 8] = hi[i]; }
  return out;
}

// ---------------------------------------------------------------------------
// Kernel 1: q/k/v projection (1x1x1 conv == per-token channel GEMV)
// ---------------------------------------------------------------------------
__global__ void qkv_kernel(const float* __restrict__ x,
                           const float* __restrict__ wq, const float* __restrict__ bq,
                           const float* __restrict__ wk, const float* __restrict__ bk,
                           const float* __restrict__ wv, const float* __restrict__ bv,
                           float* __restrict__ qf, float* __restrict__ kf,
                           half_t* __restrict__ vh) {
  int t = blockIdx.x * blockDim.x + threadIdx.x;
  if (t >= B_ * N_) return;
  int b = t >> 12, n = t & (N_ - 1);
  float xc[C_];
#pragma unroll
  for (int c = 0; c < C_; c++) xc[c] = x[((size_t)(b * C_ + c)) * N_ + n];
#pragma unroll
  for (int o = 0; o < D_; o++) {
    float aq = bq[o], ak = bk[o];
#pragma unroll
    for (int c = 0; c < C_; c++) { aq += wq[o * C_ + c] * xc[c]; ak += wk[o * C_ + c] * xc[c]; }
    qf[(size_t)t * D_ + o] = aq;
    kf[(size_t)t * D_ + o] = ak;
  }
  for (int o = 0; o < C_; o++) {
    float av = bv[o];
#pragma unroll
    for (int c = 0; c < C_; c++) av += wv[o * C_ + c] * xc[c];
    vh[((size_t)(b * C_ + o)) * N_ + n] = (half_t)av;
  }
}

// ---------------------------------------------------------------------------
// Kernel 2: landmark means (groups of 16 contiguous tokens)
// ---------------------------------------------------------------------------
__global__ void landmark_kernel(const float* __restrict__ qf, const float* __restrict__ kf,
                                float* __restrict__ qlf, float* __restrict__ klf) {
  int t = blockIdx.x * blockDim.x + threadIdx.x;
  if (t >= B_ * M_ * D_) return;
  int b = t / (M_ * D_);
  int rem = t - b * M_ * D_;
  int m = rem >> 3, o = rem & 7;
  float sq = 0.f, sk = 0.f;
  for (int l = 0; l < L_; l++) {
    size_t idx = ((size_t)b * N_ + m * L_ + l) * D_ + o;
    sq += qf[idx]; sk += kf[idx];
  }
  qlf[t] = sq * (1.f / L_);
  klf[t] = sk * (1.f / L_);
}

// ---------------------------------------------------------------------------
// Kernel 3: k1 = softmax(q . k_l^T * scale)   rows of length M=256 -> f16
// ---------------------------------------------------------------------------
__global__ void k1_kernel(const float* __restrict__ qf, const float* __restrict__ klf,
                          half_t* __restrict__ k1h) {
  int bx = blockIdx.x;                 // [0, B*N)
  int b = bx >> 12, n = bx & (N_ - 1);
  int j = threadIdx.x;                 // 256 threads: one score each
  __shared__ float red[M_];
  float qr[D_];
#pragma unroll
  for (int o = 0; o < D_; o++) qr[o] = qf[((size_t)b * N_ + n) * D_ + o];
  float s = 0.f;
#pragma unroll
  for (int o = 0; o < D_; o++) s += qr[o] * klf[((size_t)b * M_ + j) * D_ + o];
  s *= SCALE_;
  red[j] = s; __syncthreads();
  for (int off = 128; off > 0; off >>= 1) { if (j < off) red[j] = fmaxf(red[j], red[j + off]); __syncthreads(); }
  float mx = red[0]; __syncthreads();
  float e = __expf(s - mx);
  red[j] = e; __syncthreads();
  for (int off = 128; off > 0; off >>= 1) { if (j < off) red[j] += red[j + off]; __syncthreads(); }
  k1h[((size_t)b * N_ + n) * M_ + j] = (half_t)(e / red[0]);
}

// ---------------------------------------------------------------------------
// Kernel 4: k2 = softmax(q_l . k_l^T * scale)  [B,256,256] -> f32 (for NS)
// ---------------------------------------------------------------------------
__global__ void k2_kernel(const float* __restrict__ qlf, const float* __restrict__ klf,
                          float* __restrict__ k2f) {
  int bx = blockIdx.x;                 // [0, B*M)
  int b = bx >> 8, m = bx & (M_ - 1);
  int j = threadIdx.x;
  __shared__ float red[M_];
  float qr[D_];
#pragma unroll
  for (int o = 0; o < D_; o++) qr[o] = qlf[((size_t)b * M_ + m) * D_ + o];
  float s = 0.f;
#pragma unroll
  for (int o = 0; o < D_; o++) s += qr[o] * klf[((size_t)b * M_ + j) * D_ + o];
  s *= SCALE_;
  red[j] = s; __syncthreads();
  for (int off = 128; off > 0; off >>= 1) { if (j < off) red[j] = fmaxf(red[j], red[j + off]); __syncthreads(); }
  float mx = red[0]; __syncthreads();
  float e = __expf(s - mx);
  red[j] = e; __syncthreads();
  for (int off = 128; off > 0; off >>= 1) { if (j < off) red[j] += red[j + off]; __syncthreads(); }
  k2f[((size_t)b * M_ + m) * M_ + j] = e / red[0];
}

// ---------------------------------------------------------------------------
// Kernel 5: k3 = softmax over N=4096, stored TRANSPOSED as k3t[b][j][m] f16
// ---------------------------------------------------------------------------
__global__ void k3_kernel(const float* __restrict__ qlf, const float* __restrict__ kf,
                          half_t* __restrict__ k3th) {
  int bx = blockIdx.x;                 // [0, B*M)
  int b = bx >> 8, m = bx & (M_ - 1);
  int tid = threadIdx.x;               // 256 threads x 16 cols each
  __shared__ float sbuf[N_];
  __shared__ float red[256];
  float qr[D_];
#pragma unroll
  for (int o = 0; o < D_; o++) qr[o] = qlf[((size_t)b * M_ + m) * D_ + o];
  float lmax = -1e30f;
  for (int it = 0; it < 16; it++) {
    int j = it * 256 + tid;
    float s = 0.f;
#pragma unroll
    for (int o = 0; o < D_; o++) s += qr[o] * kf[((size_t)b * N_ + j) * D_ + o];
    s *= SCALE_;
    sbuf[j] = s;
    lmax = fmaxf(lmax, s);
  }
  red[tid] = lmax; __syncthreads();
  for (int off = 128; off > 0; off >>= 1) { if (tid < off) red[tid] = fmaxf(red[tid], red[tid + off]); __syncthreads(); }
  float mx = red[0]; __syncthreads();
  float lsum = 0.f;
  for (int it = 0; it < 16; it++) {
    int j = it * 256 + tid;
    float e = __expf(sbuf[j] - mx);
    sbuf[j] = e;
    lsum += e;
  }
  red[tid] = lsum; __syncthreads();
  for (int off = 128; off > 0; off >>= 1) { if (tid < off) red[tid] += red[tid + off]; __syncthreads(); }
  float inv = 1.f / red[0];
  for (int it = 0; it < 16; it++) {
    int j = it * 256 + tid;
    k3th[((size_t)b * N_ + j) * M_ + m] = (half_t)(sbuf[j] * inv);  // transposed
  }
}

// ---------------------------------------------------------------------------
// Newton-Schulz helpers (f32 for numerical stability)
// ---------------------------------------------------------------------------
__global__ void sums_kernel(const float* __restrict__ k2f,
                            float* __restrict__ cs, float* __restrict__ rs) {
  int t = blockIdx.x * blockDim.x + threadIdx.x;
  if (t >= B_ * M_) return;
  int b = t >> 8, j = t & (M_ - 1);
  const float* Kb = k2f + (size_t)b * M_ * M_;
  float c = 0.f, r = 0.f;
  for (int i = 0; i < M_; i++) { c += Kb[(size_t)i * M_ + j]; r += Kb[(size_t)j * M_ + i]; }
  cs[t] = c; rs[t] = r;
}

__global__ void alpha_kernel(const float* __restrict__ cs, const float* __restrict__ rs,
                             float* __restrict__ alpha) {
  __shared__ float red[256];
  int tid = threadIdx.x;
  red[tid] = fmaxf(cs[tid], cs[tid + 256]); __syncthreads();
  for (int off = 128; off > 0; off >>= 1) { if (tid < off) red[tid] = fmaxf(red[tid], red[tid + off]); __syncthreads(); }
  float m1 = red[0]; __syncthreads();
  red[tid] = fmaxf(rs[tid], rs[tid + 256]); __syncthreads();
  for (int off = 128; off > 0; off >>= 1) { if (tid < off) red[tid] = fmaxf(red[tid], red[tid + off]); __syncthreads(); }
  if (tid == 0) alpha[0] = m1 * red[0];
}

__global__ void v0_kernel(const float* __restrict__ k2f, const float* __restrict__ alpha,
                          float* __restrict__ V) {
  int t = blockIdx.x * blockDim.x + threadIdx.x;
  if (t >= B_ * M_ * M_) return;
  int b = t >> 16, rem = t & 65535;
  int i = rem >> 8, j = rem & (M_ - 1);
  V[t] = k2f[((size_t)b * M_ + j) * M_ + i] / alpha[0];   // K^T / alpha
}

// Y = a*I - X  (elementwise, batched 256x256)
__global__ void ew_ai_minus_kernel(const float* __restrict__ X, float* __restrict__ Y, float a) {
  int t = blockIdx.x * blockDim.x + threadIdx.x;
  if (t >= B_ * M_ * M_) return;
  int rem = t & 65535;
  int i = rem >> 8, j = rem & (M_ - 1);
  Y[t] = (i == j ? a : 0.f) - X[t];
}

// C = s * (A @ B), 256x256x256, f32, LDS-tiled
__global__ void sgemm256_kernel(const float* __restrict__ A, const float* __restrict__ Bm,
                                float* __restrict__ Cm, float s) {
  int z = blockIdx.z;
  const float* Ab = A + (size_t)z * M_ * M_;
  const float* Bb = Bm + (size_t)z * M_ * M_;
  float* Cb = Cm + (size_t)z * M_ * M_;
  __shared__ float As[16][16];
  __shared__ float Bs[16][17];
  int tx = threadIdx.x & 15, ty = threadIdx.x >> 4;
  int row = blockIdx.y * 16 + ty, col = blockIdx.x * 16 + tx;
  float acc = 0.f;
  for (int kt = 0; kt < M_; kt += 16) {
    As[ty][tx] = Ab[(size_t)row * M_ + kt + tx];
    Bs[ty][tx] = Bb[(size_t)(kt + ty) * M_ + col];
    __syncthreads();
#pragma unroll
    for (int k = 0; k < 16; k++) acc += As[ty][k] * Bs[k][tx];
    __syncthreads();
  }
  Cb[(size_t)row * M_ + col] = s * acc;
}

// invK2T[b][col j][k] = V[b][k][j]  (f32 -> f16, transposed for B-frag loads)
__global__ void cvt_inv_kernel(const float* __restrict__ V, half_t* __restrict__ invT) {
  int t = blockIdx.x * blockDim.x + threadIdx.x;
  if (t >= B_ * M_ * M_) return;
  int b = t >> 16, rem = t & 65535;
  int j = rem >> 8, k = rem & (M_ - 1);
  invT[t] = (half_t)V[((size_t)b * M_ + k) * M_ + j];
}

// ---------------------------------------------------------------------------
// WMMA GEMM: C[rows x cols] = A[rows x K] @ B[K x cols], B given transposed.
// block = 256 threads = 8 waves; each wave owns a 32x32 tile (2x2 WMMA accs);
// block tile = 128 rows x 64 cols. K must be a multiple of 32.
// mode: outH != nullptr -> store f16; else store f32 to outF.
// ---------------------------------------------------------------------------
__global__ void wmma_gemm_kernel(const half_t* __restrict__ A, int lda, long sA,
                                 const half_t* __restrict__ Bt, int ldb, long sB,
                                 half_t* __restrict__ outH, float* __restrict__ outF,
                                 int ldc, long sC, int K) {
  int z = blockIdx.z;
  const half_t* Ab = A + (size_t)z * sA;
  const half_t* Bb = Bt + (size_t)z * sB;
  int wave = threadIdx.x >> 5;
  int lane = threadIdx.x & 31;
  int row0 = blockIdx.x * 128 + (wave & 3) * 32;
  int col0 = blockIdx.y * 64 + (wave >> 2) * 32;
  v8f acc[2][2] = {};
  for (int k0 = 0; k0 < K; k0 += 32) {
    v16h a0 = load_a_frag(Ab, lda, row0, k0, lane);
    v16h a1 = load_a_frag(Ab, lda, row0 + 16, k0, lane);
    v16h b0 = load_b_frag(Bb, ldb, col0, k0, lane);
    v16h b1 = load_b_frag(Bb, ldb, col0 + 16, k0, lane);
    acc[0][0] = __builtin_amdgcn_wmma_f32_16x16x32_f16(false, a0, false, b0, (short)0, acc[0][0], false, false);
    acc[0][1] = __builtin_amdgcn_wmma_f32_16x16x32_f16(false, a0, false, b1, (short)0, acc[0][1], false, false);
    acc[1][0] = __builtin_amdgcn_wmma_f32_16x16x32_f16(false, a1, false, b0, (short)0, acc[1][0], false, false);
    acc[1][1] = __builtin_amdgcn_wmma_f32_16x16x32_f16(false, a1, false, b1, (short)0, acc[1][1], false, false);
  }
  int rsel = (lane >> 4) & 1, nsel = lane & 15;
  bool f16out = (outH != nullptr);
#pragma unroll
  for (int ti = 0; ti < 2; ti++)
#pragma unroll
    for (int tj = 0; tj < 2; tj++)
#pragma unroll
      for (int g = 0; g < 8; g++) {
        int r = row0 + ti * 16 + g + rsel * 8;
        int c = col0 + tj * 16 + nsel;
        size_t idx = (size_t)z * sC + (size_t)r * ldc + c;
        if (f16out) outH[idx] = (half_t)acc[ti][tj][g];
        else        outF[idx] = acc[ti][tj][g];
      }
}

// ---------------------------------------------------------------------------
// WMMA out-GEMM: out^T[i][c] = sum_j attn[i][j]*v[c][j]; fused gamma*acc + x,
// scattered into out[b][c][i]. A = attn (f32, converted on the fly), Bt = vh.
// ---------------------------------------------------------------------------
__global__ void wmma_out_kernel(const float* __restrict__ attn, const half_t* __restrict__ vh,
                                const float* __restrict__ x, const float* __restrict__ gamma,
                                float* __restrict__ out) {
  int z = blockIdx.z;
  const float* Ab = attn + (size_t)z * N_ * N_;
  const half_t* Bb = vh + (size_t)z * C_ * N_;
  int wave = threadIdx.x >> 5;
  int lane = threadIdx.x & 31;
  int row0 = blockIdx.x * 128 + (wave & 3) * 32;
  int col0 = (wave >> 2) * 32;          // C=64 -> 2 column waves
  v8f acc[2][2] = {};
  for (int k0 = 0; k0 < N_; k0 += 32) {
    v16h a0 = load_a_frag_f32(Ab, N_, row0, k0, lane);
    v16h a1 = load_a_frag_f32(Ab, N_, row0 + 16, k0, lane);
    v16h b0 = load_b_frag(Bb, N_, col0, k0, lane);
    v16h b1 = load_b_frag(Bb, N_, col0 + 16, k0, lane);
    acc[0][0] = __builtin_amdgcn_wmma_f32_16x16x32_f16(false, a0, false, b0, (short)0, acc[0][0], false, false);
    acc[0][1] = __builtin_amdgcn_wmma_f32_16x16x32_f16(false, a0, false, b1, (short)0, acc[0][1], false, false);
    acc[1][0] = __builtin_amdgcn_wmma_f32_16x16x32_f16(false, a1, false, b0, (short)0, acc[1][0], false, false);
    acc[1][1] = __builtin_amdgcn_wmma_f32_16x16x32_f16(false, a1, false, b1, (short)0, acc[1][1], false, false);
  }
  float g0 = gamma[0];
  int rsel = (lane >> 4) & 1, nsel = lane & 15;
#pragma unroll
  for (int ti = 0; ti < 2; ti++)
#pragma unroll
    for (int tj = 0; tj < 2; tj++)
#pragma unroll
      for (int g = 0; g < 8; g++) {
        int i = row0 + ti * 16 + g + rsel * 8;   // token index
        int c = col0 + tj * 16 + nsel;           // channel index
        size_t idx = (size_t)z * C_ * N_ + (size_t)c * N_ + i;
        out[idx] = g0 * acc[ti][tj][g] + x[idx];
      }
}

// ---------------------------------------------------------------------------
// Host driver
// ---------------------------------------------------------------------------
extern "C" void kernel_launch(void* const* d_in, const int* in_sizes, int n_in,
                              void* d_out, int out_size, void* d_ws, size_t ws_size,
                              hipStream_t stream) {
  const float* x     = (const float*)d_in[0];
  const float* wq    = (const float*)d_in[1];
  const float* bq    = (const float*)d_in[2];
  const float* wk    = (const float*)d_in[3];
  const float* bk    = (const float*)d_in[4];
  const float* wv    = (const float*)d_in[5];
  const float* bv    = (const float*)d_in[6];
  const float* gamma = (const float*)d_in[7];

  float* out_f  = (float*)d_out;                       // [B, C, N]
  float* attn_f = out_f + (size_t)B_ * C_ * N_;        // [B, N, N]

  // workspace sub-allocation (256B aligned)
  size_t off = 0;
  auto sub = [&](size_t bytes) -> char* {
    char* p = (char*)d_ws + off;
    off += (bytes + 255) & ~(size_t)255;
    return p;
  };
  float*  qf     = (float*)sub((size_t)B_ * N_ * D_ * 4);
  float*  kf     = (float*)sub((size_t)B_ * N_ * D_ * 4);
  float*  qlf    = (float*)sub((size_t)B_ * M_ * D_ * 4);
  float*  klf    = (float*)sub((size_t)B_ * M_ * D_ * 4);
  half_t* vh     = (half_t*)sub((size_t)B_ * C_ * N_ * 2);
  half_t* k1h    = (half_t*)sub((size_t)B_ * N_ * M_ * 2);
  float*  k2f    = (float*)sub((size_t)B_ * M_ * M_ * 4);
  half_t* k3th   = (half_t*)sub((size_t)B_ * N_ * M_ * 2);
  float*  cs     = (float*)sub((size_t)B_ * M_ * 4);
  float*  rs     = (float*)sub((size_t)B_ * M_ * 4);
  float*  alpha  = (float*)sub(256);
  float*  Vf     = (float*)sub((size_t)B_ * M_ * M_ * 4);
  float*  V2f    = (float*)sub((size_t)B_ * M_ * M_ * 4);
  float*  KVf    = (float*)sub((size_t)B_ * M_ * M_ * 4);
  float*  T1f    = (float*)sub((size_t)B_ * M_ * M_ * 4);
  float*  T2f    = (float*)sub((size_t)B_ * M_ * M_ * 4);
  half_t* invT   = (half_t*)sub((size_t)B_ * M_ * M_ * 2);
  half_t* S1h    = (half_t*)sub((size_t)B_ * N_ * M_ * 2);
  (void)ws_size; (void)in_sizes; (void)n_in; (void)out_size;

  // 1) projections
  qkv_kernel<<<(B_ * N_ + 255) / 256, 256, 0, stream>>>(x, wq, bq, wk, bk, wv, bv, qf, kf, vh);
  // 2) landmarks
  landmark_kernel<<<(B_ * M_ * D_ + 255) / 256, 256, 0, stream>>>(qf, kf, qlf, klf);
  // 3-5) softmax attention factors
  k1_kernel<<<B_ * N_, 256, 0, stream>>>(qf, klf, k1h);
  k2_kernel<<<B_ * M_, 256, 0, stream>>>(qlf, klf, k2f);
  k3_kernel<<<B_ * M_, 256, 0, stream>>>(qlf, kf, k3th);
  // 6-8) Newton-Schulz init: V0 = K^T / (max colsum * max rowsum)
  sums_kernel<<<(B_ * M_ + 255) / 256, 256, 0, stream>>>(k2f, cs, rs);
  alpha_kernel<<<1, 256, 0, stream>>>(cs, rs, alpha);
  v0_kernel<<<(B_ * M_ * M_ + 255) / 256, 256, 0, stream>>>(k2f, alpha, Vf);
  // 9) 6 Newton-Schulz iterations (f32)
  dim3 gsg(16, 16, B_);
  int ewg = (B_ * M_ * M_ + 255) / 256;
  float* cur = Vf;
  float* spare = V2f;
  for (int it = 0; it < NS_ITERS; it++) {
    sgemm256_kernel<<<gsg, 256, 0, stream>>>(k2f, cur, KVf, 1.0f);       // KV = K@V
    ew_ai_minus_kernel<<<ewg, 256, 0, stream>>>(KVf, T1f, 7.0f);         // 7I - KV
    sgemm256_kernel<<<gsg, 256, 0, stream>>>(KVf, T1f, T2f, 1.0f);       // KV@(7I-KV)
    ew_ai_minus_kernel<<<ewg, 256, 0, stream>>>(T2f, T1f, 15.0f);        // 15I - ...
    sgemm256_kernel<<<gsg, 256, 0, stream>>>(KVf, T1f, T2f, 1.0f);       // KV@(15I-...)
    ew_ai_minus_kernel<<<ewg, 256, 0, stream>>>(T2f, T1f, 13.0f);        // 13I - ...
    sgemm256_kernel<<<gsg, 256, 0, stream>>>(cur, T1f, spare, 0.25f);    // V = 0.25*V@(...)
    float* tmp = cur; cur = spare; spare = tmp;
  }
  // 10) f16 transpose of inverse
  cvt_inv_kernel<<<ewg, 256, 0, stream>>>(cur, invT);
  // 11) S1 = k1 @ invK2   [N x 256] (WMMA, f16 out)
  {
    dim3 grid(N_ / 128, M_ / 64, B_);
    wmma_gemm_kernel<<<grid, 256, 0, stream>>>(k1h, M_, (long)N_ * M_,
                                               invT, M_, (long)M_ * M_,
                                               S1h, nullptr, M_, (long)N_ * M_, M_);
  }
  // 12) attn = S1 @ k3    [N x N] (WMMA, f32 out to d_out)
  {
    dim3 grid(N_ / 128, N_ / 64, B_);
    wmma_gemm_kernel<<<grid, 256, 0, stream>>>(S1h, M_, (long)N_ * M_,
                                               k3th, M_, (long)N_ * M_,
                                               nullptr, attn_f, N_, (long)N_ * N_, M_);
  }
  // 13) out = gamma * (v . attn^T) + x   (WMMA, K = 4096)
  {
    dim3 grid(N_ / 128, 1, B_);
    wmma_out_kernel<<<grid, 256, 0, stream>>>(attn_f, vh, x, gamma, out_f);
  }
}
